// GDConv2d_23141283791600
// MI455X (gfx1250) — matrix operand order repaired and checked by reading, not compiled, not run on hardware
//
#include <hip/hip_runtime.h>

// ---------------- problem dims ----------------
constexpr int NB    = 4;      // batch
constexpr int C_IN  = 128;
constexpr int C_OUT = 128;
constexpr int IMG   = 128;    // H == W
constexpr int KK    = 9;      // 3x3 taps
constexpr int KDIM  = C_IN * KK;      // 1152 = GEMM K
constexpr int KSTEP = 32;             // bf16 WMMA K per step
constexpr int NKS   = KDIM / KSTEP;   // 36

// ---------------- vector types ----------------
typedef __attribute__((ext_vector_type(16))) __bf16        v16bf;
typedef __attribute__((ext_vector_type(8)))  float         v8f;
typedef __attribute__((ext_vector_type(8)))  unsigned int  v8u;
typedef __attribute__((ext_vector_type(4)))  unsigned int  v4u;
typedef __attribute__((ext_vector_type(4)))  unsigned int  u32x4;
typedef __attribute__((ext_vector_type(8)))  int           i32x8;
typedef __attribute__((ext_vector_type(4)))  int           i32x4;

static __device__ __forceinline__ unsigned int pk_bf16(float a, float b) {
    // round-to-nearest-even f32 -> bf16, packed pair (lo = a, hi = b)
    unsigned int ua = __builtin_bit_cast(unsigned int, a);
    unsigned int ub = __builtin_bit_cast(unsigned int, b);
    ua = (ua + 0x7FFFu + ((ua >> 16) & 1u)) >> 16;
    ub = (ub + 0x7FFFu + ((ub >> 16) & 1u)) >> 16;
    return (ua & 0xFFFFu) | (ub << 16);
}
static __device__ __forceinline__ v16bf as_bf16x16(v8u u) {
    return __builtin_bit_cast(v16bf, u);
}

// ---------------- TDM: 1-D global -> LDS DMA (ndw dwords) ----------------
// D# group0: [0]=count=1 | [1]=lds_addr | [2]=global_addr[31:0]
//            [3]=global_addr[56:32] | type=2<<30
// D# group1: data_size=4B; tensor_dim0 = tile_dim0 = ndw; stride = ndw; 1 row.
static __device__ __forceinline__ void tdm_load_1d(unsigned int lds_off,
                                                   const void* gptr,
                                                   unsigned int ndw) {
    unsigned long long ga = (unsigned long long)(size_t)gptr;
    u32x4 g0 = {0u, 0u, 0u, 0u};
    g0[0] = 1u;                                                   // count=1 (user D#)
    g0[1] = lds_off;                                              // lds_addr
    g0[2] = (unsigned int)(ga & 0xFFFFFFFFull);                   // global_addr[31:0]
    g0[3] = (unsigned int)((ga >> 32) & 0x01FFFFFFull) | (2u << 30); // addr[56:32] | type=2
    i32x8 g1 = {0, 0, 0, 0, 0, 0, 0, 0};
    g1[0] = (int)(2u << 16);          // data_size = 4 bytes; wg_mask = 0 (not in cluster)
    g1[1] = (int)((ndw & 0xFFFFu) << 16);   // tensor_dim0[15:0]  (bits 63:48)
    g1[2] = (int)(1u << 16);          // tensor_dim0[31:16]=0, tensor_dim1=1
    g1[3] = (int)((ndw & 0xFFFFu) << 16);   // tile_dim0 (bits 127:112)
    g1[4] = 0;                        // tile_dim1 = 0 (unused), tile_dim2 = 0
    g1[5] = (int)ndw;                 // tensor_dim0_stride[31:0]
    i32x4 z = {0, 0, 0, 0};
#if defined(__clang_major__) && (__clang_major__ >= 23)
    i32x8 z8 = {0, 0, 0, 0, 0, 0, 0, 0};
    __builtin_amdgcn_tensor_load_to_lds(g0, g1, z, z, z8, 0);
#else
    __builtin_amdgcn_tensor_load_to_lds(g0, g1, z, z, 0);
#endif
}

// ---------------- workspace layout (bytes) ----------------
constexpr size_t SZ_INP_NHWC  = (size_t)NB * IMG * IMG * C_IN * 4;  // 32 MB f32
constexpr size_t SZ_FEAT_NHWC = (size_t)NB * IMG * IMG * C_IN * 2;  // 16 MB bf16
constexpr size_t SZ_BDC_SW    = (size_t)NKS * 8 * 32 * 16 * 2;      // 288 KB bf16
constexpr size_t SZ_BOFF_SW   = (size_t)NKS * 2 * 32 * 16 * 2;      // 72 KB bf16
constexpr size_t OFF_INP   = 0;
constexpr size_t OFF_FEAT  = OFF_INP + SZ_INP_NHWC;
constexpr size_t OFF_BDC   = OFF_FEAT + SZ_FEAT_NHWC;
constexpr size_t OFF_BOFF  = OFF_BDC + SZ_BDC_SW;
constexpr size_t OFF_OM    = OFF_BOFF + SZ_BOFF_SW;                 // offmask f32 [B][H][W][32]

// ============================================================
// 1) NCHW f32 -> NHWC f32 (tiled transpose through LDS)
// ============================================================
__global__ __launch_bounds__(256)
void k_nchw_to_nhwc_f32(const float* __restrict__ src, float* __restrict__ dst) {
    int bid = blockIdx.x;
    int wc  = bid % (IMG / 64);
    int t2  = bid / (IMG / 64);
    int h   = t2 % IMG;
    int bb  = t2 / IMG;
    int w0  = wc * 64;
    int tid = threadIdx.x;
    __shared__ float tile[4 * 64];
    for (int c0 = 0; c0 < C_IN; c0 += 4) {
        int ty = tid >> 6, tx = tid & 63;
        tile[ty * 64 + tx] = src[(((size_t)(bb * C_IN + c0 + ty) * IMG + h) * IMG) + w0 + tx];
        __syncthreads();
        int wy = tid >> 2, cy = tid & 3;
        dst[((size_t)((bb * IMG + h) * IMG) + w0 + wy) * C_IN + c0 + cy] = tile[cy * 64 + wy];
        __syncthreads();
    }
}

// ============================================================
// 2) NCHW f32 -> NHWC bf16 (packed pairs as uint)
// ============================================================
__global__ __launch_bounds__(256)
void k_nchw_to_nhwc_bf16(const float* __restrict__ src, unsigned int* __restrict__ dst) {
    int bid = blockIdx.x;
    int wc  = bid % (IMG / 32);
    int t2  = bid / (IMG / 32);
    int h   = t2 % IMG;
    int bb  = t2 / IMG;
    int w0  = wc * 32;
    int tid = threadIdx.x;
    __shared__ float tile[8 * 32];
    for (int c0 = 0; c0 < C_IN; c0 += 8) {
        int ty = tid >> 5, tx = tid & 31;
        tile[ty * 32 + tx] = src[(((size_t)(bb * C_IN + c0 + ty) * IMG + h) * IMG) + w0 + tx];
        __syncthreads();
        if (tid < 128) {
            int wy = tid >> 2, cp = tid & 3;
            float f0 = tile[(cp * 2 + 0) * 32 + wy];
            float f1 = tile[(cp * 2 + 1) * 32 + wy];
            size_t ei = ((size_t)((bb * IMG + h) * IMG) + w0 + wy) * C_IN + c0 + cp * 2;
            dst[ei >> 1] = pk_bf16(f0, f1);
        }
        __syncthreads();
    }
}

// ============================================================
// 3) Weight swizzle: [N][C_IN][3][3] f32 -> per-lane WMMA-B bf16 fragments
//    dst[(ks*ntiles + nt)*32 + lane][8 uints]; k = tap*C_IN + c (tap-major)
// ============================================================
__global__ __launch_bounds__(256)
void k_swizzle_w(const float* __restrict__ w, int nvalid, int ntiles,
                 unsigned int* __restrict__ dst) {
    int idx   = blockIdx.x * blockDim.x + threadIdx.x;
    int total = NKS * ntiles * 32 * 8;
    if (idx >= total) return;
    int j2   = idx & 7;
    int lane = (idx >> 3) & 31;
    int nt   = (idx >> 8) % ntiles;
    int ks   = (idx >> 8) / ntiles;
    int n    = nt * 16 + (lane & 15);
    int kb   = ks * KSTEP + ((lane >> 4) << 3) + ((j2 & 3) << 1) + ((j2 >> 2) << 4);
    float f0 = 0.f, f1 = 0.f;
    if (n < nvalid) {
        int t0 = kb / C_IN,       c0 = kb % C_IN;
        int t1 = (kb + 1) / C_IN, c1 = (kb + 1) % C_IN;
        f0 = w[((size_t)n * C_IN + c0) * KK + t0];
        f1 = w[((size_t)n * C_IN + c1) * KK + t1];
    }
    dst[idx] = pk_bf16(f0, f1);
}

// ============================================================
// 4) Offset conv as WMMA GEMM:  M=128, N=32 (27 valid), K=1152
//    B fragments staged LDS-side via TDM, double-buffered.
// ============================================================
__global__ __launch_bounds__(256)
void k_offset_gemm(const unsigned int* __restrict__ feat_nhwc,
                   const unsigned int* __restrict__ boff_sw,
                   const float* __restrict__ b_off,
                   float* __restrict__ offmask) {
    int bid  = blockIdx.x;
    int h    = bid % IMG;
    int bb   = bid / IMG;
    int tid  = threadIdx.x;
    int lane = tid & 31;
    int wave = tid >> 5;

    __shared__ unsigned int Atile[128 * 16];   // 8 KB (fragment-permuted K order)
    __shared__ unsigned int Btile[2][512];     // 2 x 2 KB, TDM double buffer

    constexpr unsigned int BBLK = 2 * 32 * 8;  // dwords per K-step B block (512)

    if (wave == 0)
        tdm_load_1d((unsigned int)(size_t)&Btile[0][0], boff_sw, BBLK);

    v8f acc0 = {0.f, 0.f, 0.f, 0.f, 0.f, 0.f, 0.f, 0.f};
    v8f acc1 = acc0;

    for (int ks = 0; ks < NKS; ++ks) {
        int tap = ks >> 2;
        int cb  = (ks & 3) * KSTEP;
        int dy  = tap / 3 - 1, dx = tap % 3 - 1;
        // ---- build A tile: thread t -> row m = t/2, half = t&1 ----
        int m    = tid >> 1;
        int half = tid & 1;
        int iy   = h + dy;
        int ix   = m + dx;
        v4u c0 = {0u, 0u, 0u, 0u};
        v4u c1 = {0u, 0u, 0u, 0u};
        if (iy >= 0 && iy < IMG && ix >= 0 && ix < IMG) {
            const unsigned int* p =
                feat_nhwc + (size_t)((bb * IMG + iy) * IMG + ix) * (C_IN / 2);
            int cA = cb + half * 8;               // fragment-permuted chunks: +0 and +16
            c0 = *(const v4u*)(p + (cA >> 1));
            c1 = *(const v4u*)(p + ((cA + 16) >> 1));
        }
        v8u comb = __builtin_shufflevector(c0, c1, 0, 1, 2, 3, 4, 5, 6, 7);
        *(v8u*)&Atile[m * 16 + half * 8] = comb;

        // ---- wave0: kick next B-block DMA, ensure current block landed ----
        if (wave == 0) {
            if (ks + 1 < NKS) {
                tdm_load_1d((unsigned int)(size_t)&Btile[(ks + 1) & 1][0],
                            boff_sw + (size_t)(ks + 1) * BBLK, BBLK);
                __builtin_amdgcn_s_wait_tensorcnt(1);
            } else {
                __builtin_amdgcn_s_wait_tensorcnt(0);
            }
        }
        __syncthreads();

        // ---- WMMA: wave owns M-strip [wave*16, +16), N tiles 0..1 ----
        int m0 = wave * 16;
        const unsigned int* bt = &Btile[ks & 1][0];
        v16bf a  = as_bf16x16(*(const v8u*)&Atile[(m0 + (lane & 15)) * 16 + (lane >> 4) * 8]);
        v16bf b0 = as_bf16x16(*(const v8u*)&bt[(0 * 32 + lane) * 8]);
        v16bf b1 = as_bf16x16(*(const v8u*)&bt[(1 * 32 + lane) * 8]);
        acc0 = __builtin_amdgcn_wmma_f32_16x16x32_bf16(false, a, false, b0, (short)0, acc0, false, false);
        acc1 = __builtin_amdgcn_wmma_f32_16x16x32_bf16(false, a, false, b1, (short)0, acc1, false, false);
        __syncthreads();
    }

    // ---- epilogue: VGPR r, lane l -> M = m0 + r + (l>=16)*8, N = nt*16 + l%16
    int mrow = wave * 16 + ((lane >> 4) << 3);
    float* outp = offmask + (size_t)((bb * IMG + h) * IMG) * 32;
    int n0 = lane & 15;        // 0..15  -> raw offsets
    int n1 = 16 + n0;          // 16,17 raw; 18..26 sigmoid; rest dropped
    float bia0 = b_off[n0];
    float bia1 = (n1 < 27) ? b_off[n1] : 0.f;
#pragma unroll
    for (int r = 0; r < 8; ++r) {
        int w = mrow + r;
        outp[(size_t)w * 32 + n0] = acc0[r] + bia0;
        float v1 = acc1[r] + bia1;
        if (n1 < 18)      outp[(size_t)w * 32 + n1] = v1;
        else if (n1 < 27) outp[(size_t)w * 32 + n1] = 1.f / (1.f + __expf(-v1));
    }
}

// ============================================================
// 5) Fused deformable-sampling + WMMA GEMM
//    M=128, N=128, K=1152; B fragments via TDM double buffer.
// ============================================================
__global__ __launch_bounds__(256)
void k_deform_gemm(const float* __restrict__ inp_nhwc,
                   const unsigned int* __restrict__ bdc_sw,
                   const float* __restrict__ offmask,
                   const float* __restrict__ b_dc,
                   float* __restrict__ out) {
    int bid  = blockIdx.x;
    int h    = bid % IMG;
    int bb   = bid / IMG;
    int tid  = threadIdx.x;
    int lane = tid & 31;
    int wave = tid >> 5;

    __shared__ unsigned int Atile[128 * 16];   // 8 KB
    __shared__ unsigned int Btile[2][2048];    // 2 x 8 KB, TDM double buffer
    __shared__ float SW[KK][128][4];           // 18 KB  corner weights (x mask, zeroed OOB)
    __shared__ int   SI[KK][128][4];           // 18 KB  corner pixel-base element indices

    constexpr unsigned int BBLK = 8 * 32 * 8;  // dwords per K-step B block (2048)

    if (wave == 0)
        tdm_load_1d((unsigned int)(size_t)&Btile[0][0], bdc_sw, BBLK);

    // ---- precompute bilinear corners for all 128 positions x 9 taps ----
    for (int t = tid; t < KK * 128; t += 256) {
        int tap = t / 128;
        int m   = t % 128;
        const float* om = offmask + (size_t)((bb * IMG + h) * IMG + m) * 32;
        float offy = om[2 * tap], offx = om[2 * tap + 1], mk = om[18 + tap];
        int dy = tap / 3 - 1, dx = tap % 3 - 1;
        float py = (float)(h + dy) + offy;
        float px = (float)(m + dx) + offx;
        float y0f = floorf(py), x0f = floorf(px);
        float ly = py - y0f, lx = px - x0f;
        int y0 = (int)y0f, x0 = (int)x0f;
        float ws0 = (1.f - ly) * (1.f - lx) * mk;
        float ws1 = (1.f - ly) * lx * mk;
        float ws2 = ly * (1.f - lx) * mk;
        float ws3 = ly * lx * mk;
        float wv[4] = {ws0, ws1, ws2, ws3};
#pragma unroll
        for (int c = 0; c < 4; ++c) {
            int yy = y0 + (c >> 1);
            int xx = x0 + (c & 1);
            bool valid = (yy >= 0) && (yy < IMG) && (xx >= 0) && (xx < IMG);
            int yc = yy < 0 ? 0 : (yy > IMG - 1 ? IMG - 1 : yy);
            int xc = xx < 0 ? 0 : (xx > IMG - 1 ? IMG - 1 : xx);
            SI[tap][m][c] = ((bb * IMG + yc) * IMG + xc) * C_IN;
            SW[tap][m][c] = valid ? wv[c] : 0.f;
        }
    }
    __syncthreads();

    v8f vz = {0.f, 0.f, 0.f, 0.f, 0.f, 0.f, 0.f, 0.f};
    v8f acc[8];
#pragma unroll
    for (int i = 0; i < 8; ++i) acc[i] = vz;

    for (int ks = 0; ks < NKS; ++ks) {
        int tap = ks >> 2;
        int cb  = (ks & 3) * KSTEP;
        // ---- build bf16 A tile with on-the-fly bilinear sampling ----
        int m    = tid >> 1;
        int half = tid & 1;
        float w0 = SW[tap][m][0], w1 = SW[tap][m][1], w2 = SW[tap][m][2], w3 = SW[tap][m][3];
        int   i0 = SI[tap][m][0] + cb, i1 = SI[tap][m][1] + cb;
        int   i2 = SI[tap][m][2] + cb, i3 = SI[tap][m][3] + cb;
        v8u packed;
#pragma unroll
        for (int r = 0; r < 2; ++r) {
            int c = half * 8 + r * 16;           // fragment-permuted channel chunk
            v8f a0 = *(const v8f*)(inp_nhwc + i0 + c);
            v8f a1 = *(const v8f*)(inp_nhwc + i1 + c);
            v8f a2 = *(const v8f*)(inp_nhwc + i2 + c);
            v8f a3 = *(const v8f*)(inp_nhwc + i3 + c);
#pragma unroll
            for (int j = 0; j < 4; ++j) {
                float e0 = w0 * a0[2 * j] + w1 * a1[2 * j] + w2 * a2[2 * j] + w3 * a3[2 * j];
                float e1 = w0 * a0[2 * j + 1] + w1 * a1[2 * j + 1] + w2 * a2[2 * j + 1] + w3 * a3[2 * j + 1];
                packed[r * 4 + j] = pk_bf16(e0, e1);
            }
        }
        *(v8u*)&Atile[m * 16 + half * 8] = packed;

        // ---- wave0: kick next B-block DMA, ensure current block landed ----
        if (wave == 0) {
            if (ks + 1 < NKS) {
                tdm_load_1d((unsigned int)(size_t)&Btile[(ks + 1) & 1][0],
                            bdc_sw + (size_t)(ks + 1) * BBLK, BBLK);
                __builtin_amdgcn_s_wait_tensorcnt(1);
            } else {
                __builtin_amdgcn_s_wait_tensorcnt(0);
            }
        }
        __syncthreads();

        // ---- WMMA: each wave: M-strip = wave*16, all 8 N tiles ----
        int m0 = wave * 16;
        v16bf a = as_bf16x16(*(const v8u*)&Atile[(m0 + (lane & 15)) * 16 + (lane >> 4) * 8]);
        const unsigned int* bt = &Btile[ks & 1][0];
#pragma unroll
        for (int nt = 0; nt < 8; ++nt) {
            v16bf bq = as_bf16x16(*(const v8u*)&bt[(nt * 32 + lane) * 8]);
            acc[nt] = __builtin_amdgcn_wmma_f32_16x16x32_bf16(false, a, false, bq, (short)0,
                                                              acc[nt], false, false);
        }
        __syncthreads();
    }

    // ---- epilogue: per-lane 8 consecutive w positions -> contiguous 32B stores ----
    int mrow = wave * 16 + ((lane >> 4) << 3);
#pragma unroll
    for (int nt = 0; nt < 8; ++nt) {
        int o = nt * 16 + (lane & 15);
        float bias = b_dc[o];
        float* dst = out + ((size_t)(bb * C_OUT + o) * IMG + h) * IMG + mrow;
        v8f v = acc[nt];
#pragma unroll
        for (int r = 0; r < 8; ++r) dst[r] = v[r] + bias;
    }
}

// ============================================================
// launch
// ============================================================
extern "C" void kernel_launch(void* const* d_in, const int* in_sizes, int n_in,
                              void* d_out, int out_size, void* d_ws, size_t ws_size,
                              hipStream_t stream) {
    const float* inp   = (const float*)d_in[0];
    const float* feat  = (const float*)d_in[1];
    const float* w_off = (const float*)d_in[2];
    const float* b_off = (const float*)d_in[3];
    const float* w_dc  = (const float*)d_in[4];
    const float* b_dc  = (const float*)d_in[5];
    float* out = (float*)d_out;

    char* ws = (char*)d_ws;
    float*        inp_nhwc  = (float*)(ws + OFF_INP);
    unsigned int* feat_nhwc = (unsigned int*)(ws + OFF_FEAT);
    unsigned int* bdc_sw    = (unsigned int*)(ws + OFF_BDC);
    unsigned int* boff_sw   = (unsigned int*)(ws + OFF_BOFF);
    float*        offmask   = (float*)(ws + OFF_OM);

    k_nchw_to_nhwc_f32 <<<NB * IMG * (IMG / 64), 256, 0, stream>>>(inp, inp_nhwc);
    k_nchw_to_nhwc_bf16<<<NB * IMG * (IMG / 32), 256, 0, stream>>>(feat, feat_nhwc);

    int tot_dc  = NKS * 8 * 32 * 8;
    int tot_off = NKS * 2 * 32 * 8;
    k_swizzle_w<<<(tot_dc  + 255) / 256, 256, 0, stream>>>(w_dc,  C_OUT, 8, bdc_sw);
    k_swizzle_w<<<(tot_off + 255) / 256, 256, 0, stream>>>(w_off, 27,    2, boff_sw);

    k_offset_gemm<<<NB * IMG, 256, 0, stream>>>(feat_nhwc, boff_sw, b_off, offmask);
    k_deform_gemm<<<NB * IMG, 256, 0, stream>>>(inp_nhwc, bdc_sw, offmask, b_dc, out);
}
